// CRF_39273180955615
// MI455X (gfx1250) — compile-verified, matrix operand order checked by reading, not run on hardware
//
#include <hip/hip_runtime.h>
#include <hip/hip_bf16.h>

// ---------------------------------------------------------------------------
// CRF log-likelihood, fused: score = X@W via v_wmma_f32_16x16x32_bf16 into
// LDS, then forward log-recurrence (lane = label) + gold-path scoring.
// B=8192, L=32, F=128, K=26 (padded to 32 cols).
// ---------------------------------------------------------------------------

typedef __attribute__((ext_vector_type(16))) __bf16 v16bf;
typedef __attribute__((ext_vector_type(8)))  float  v8f;
typedef __attribute__((ext_vector_type(4)))  float  v4f;

#define CRF_B 8192
#define CRF_L 32
#define CRF_F 128
#define CRF_K 26
#define KPAD  32
#define BT    8              // batch elements per workgroup
#define ROWS  (BT * CRF_L)   // 256 score rows per workgroup
#define WBSW  (2 * 4 * 32 * 16)  // swizzled W: [ntile][kstep][lane][16] bf16

__device__ __forceinline__ float bcast_lane(float v, int srcLane) {
    return __builtin_bit_cast(float,
        __builtin_amdgcn_readlane(__builtin_bit_cast(int, v), srcLane));
}

__global__ void __launch_bounds__(256)
crf_fused_kernel(const float* __restrict__ X,   // [B, L, F]
                 const int*   __restrict__ Y,   // [B, L]
                 const float* __restrict__ W,   // [F, K]
                 const float* __restrict__ T,   // [K, K]
                 float*       __restrict__ part) // [gridDim.x] partial sums
{
    __shared__ float  sScore[ROWS * KPAD];   // 32 KB
    __shared__ __bf16 sWB[WBSW];             // 8 KB, WMMA-B register layout
    __shared__ float  sPart[BT];

    const int tid  = threadIdx.x;
    const int wave = tid >> 5;
    const int lane = tid & 31;
    const int half = lane >> 4;      // 0 or 1
    const int mn   = lane & 15;      // M (for A/C) or N (for B/C)

    // ---- stage W into LDS pre-swizzled into the 16-bit B-matrix layout ----
    // Element j of lane (n, half) at kstep ks holds W[k][col] with
    //   kb = 32*ks + 8*half,  k = kb + j + (j>=8 ? 8 : 0),  col = 16*nt + n
    // so each lane's 16-element fragment is contiguous (2x ds_load_b128).
    for (int idx = tid; idx < WBSW; idx += 256) {
        const int j   = idx & 15;
        const int ln  = (idx >> 4) & 31;
        const int ks  = (idx >> 9) & 3;
        const int nt  = (idx >> 11) & 1;
        const int k   = 32 * ks + 8 * (ln >> 4) + j + ((j >= 8) ? 8 : 0);
        const int col = 16 * nt + (ln & 15);
        sWB[idx] = (col < CRF_K) ? (__bf16)W[k * CRF_K + col] : (__bf16)0.0f;
    }
    __syncthreads();

    // ---- preload B fragments into registers (shared by both row tiles) ----
    v16bf B0[4], B1[4];
#pragma unroll
    for (int ks = 0; ks < 4; ++ks) {
        B0[ks] = *reinterpret_cast<const v16bf*>(&sWB[((0 * 4 + ks) * 32 + lane) * 16]);
        B1[ks] = *reinterpret_cast<const v16bf*>(&sWB[((1 * 4 + ks) * 32 + lane) * 16]);
    }

    // ---- Phase 1: GEMM. Each wave does row tiles rt = wave, wave+8 ----
    const long rowBase = (long)blockIdx.x * ROWS;
#pragma unroll
    for (int rto = 0; rto < 2; ++rto) {
        const int rt = wave + rto * 8;
        const float* xr = X + (rowBase + rt * 16 + mn) * CRF_F;

        v8f c0 = {}; // cols 0..15
        v8f c1 = {}; // cols 16..31
#pragma unroll
        for (int ks = 0; ks < 4; ++ks) {
            const int kb = ks * 32 + half * 8;
            // A layout (16x32 bf16): lane (m, half) holds K in
            // [kb, kb+8) and [kb+16, kb+24), packed pairs in increasing K.
            const v4f* p = (const v4f*)(xr + kb);
            const v4f* q = (const v4f*)(xr + kb + 16);
            v4f p0 = __builtin_nontemporal_load(p);
            v4f p1 = __builtin_nontemporal_load(p + 1);
            v4f q0 = __builtin_nontemporal_load(q);
            v4f q1 = __builtin_nontemporal_load(q + 1);
            v16bf a;
            a[0]=(__bf16)p0[0]; a[1]=(__bf16)p0[1]; a[2]=(__bf16)p0[2]; a[3]=(__bf16)p0[3];
            a[4]=(__bf16)p1[0]; a[5]=(__bf16)p1[1]; a[6]=(__bf16)p1[2]; a[7]=(__bf16)p1[3];
            a[8]=(__bf16)q0[0]; a[9]=(__bf16)q0[1]; a[10]=(__bf16)q0[2]; a[11]=(__bf16)q0[3];
            a[12]=(__bf16)q1[0]; a[13]=(__bf16)q1[1]; a[14]=(__bf16)q1[2]; a[15]=(__bf16)q1[3];

            c0 = __builtin_amdgcn_wmma_f32_16x16x32_bf16(
                     false, a, false, B0[ks], (short)0, c0, false, false);
            c1 = __builtin_amdgcn_wmma_f32_16x16x32_bf16(
                     false, a, false, B1[ks], (short)0, c1, false, false);
        }
        // C layout: VGPR r -> row (r + 8*half), col mn
#pragma unroll
        for (int r = 0; r < 8; ++r) {
            const int row = rt * 16 + r + 8 * half;
            sScore[row * KPAD + mn]      = c0[r];
            sScore[row * KPAD + 16 + mn] = c1[r];
        }
    }
    __syncthreads();

    // ---- Phase 2: forward recurrence. Wave w handles batch element w ----
    const int e = wave;                       // element within block
    const long b = (long)blockIdx.x * BT + e; // global batch index

    // lane = label column; preload T column T[k][lane]
    float Tc[CRF_K];
#pragma unroll
    for (int k = 0; k < CRF_K; ++k)
        Tc[k] = (lane < CRF_K) ? T[k * CRF_K + lane] : 0.0f;

    const float* sc = &sScore[e * CRF_L * KPAD];

    float l = 0.0f;   // forward message for this lane's label
    for (int i = 0; i < CRF_L - 1; ++i) {
        const float v = l + sc[i * KPAD + lane];   // l[i] + score[i], per label
        float t[CRF_K];
        float m = -3.0e38f;
#pragma unroll
        for (int k = 0; k < CRF_K; ++k) {
            t[k] = bcast_lane(v, k) + Tc[k];
            m = fmaxf(m, t[k]);
        }
        float s = 0.0f;
#pragma unroll
        for (int k = 0; k < CRF_K; ++k)
            s += __expf(t[k] - m);
        l = m + __logf(s);
    }

    // logZ = logsumexp_k( l[L-1] + score[L-1] )  (uniform across wave)
    const float vf = l + sc[(CRF_L - 1) * KPAD + lane];
    float mz = -3.0e38f;
#pragma unroll
    for (int k = 0; k < CRF_K; ++k)
        mz = fmaxf(mz, bcast_lane(vf, k));
    float sz = 0.0f;
#pragma unroll
    for (int k = 0; k < CRF_K; ++k)
        sz += __expf(bcast_lane(vf, k) - mz);
    const float logZ = mz + __logf(sz);

    // gold path: lane = position i (L == 32 == wave width)
    const int yi = Y[b * CRF_L + lane];
    float g = sc[lane * KPAD + yi];                       // emission
    const int yn = __shfl_down(yi, 1, 32);
    if (lane < CRF_L - 1) g += T[yi * CRF_K + yn];        // transition
#pragma unroll
    for (int off = 16; off > 0; off >>= 1)
        g += __shfl_xor(g, off, 32);

    if (lane == 0) sPart[wave] = g - logZ;
    __syncthreads();

    if (tid == 0) {
        float acc = 0.0f;
#pragma unroll
        for (int w = 0; w < BT; ++w) acc += sPart[w];
        part[blockIdx.x] = acc;
    }
}

__global__ void __launch_bounds__(256)
crf_reduce_kernel(const float* __restrict__ part, float* __restrict__ out, int n)
{
    __shared__ float sm[256];
    float s = 0.0f;
    for (int i = threadIdx.x; i < n; i += 256) s += part[i];
    sm[threadIdx.x] = s;
    __syncthreads();
    for (int off = 128; off > 0; off >>= 1) {
        if (threadIdx.x < off) sm[threadIdx.x] += sm[threadIdx.x + off];
        __syncthreads();
    }
    if (threadIdx.x == 0) out[0] = sm[0] * (1.0f / (float)CRF_B);
}

extern "C" void kernel_launch(void* const* d_in, const int* in_sizes, int n_in,
                              void* d_out, int out_size, void* d_ws, size_t ws_size,
                              hipStream_t stream) {
    const float* X = (const float*)d_in[0];
    const int*   Y = (const int*)  d_in[1];
    const float* W = (const float*)d_in[2];
    const float* T = (const float*)d_in[3];
    float* out  = (float*)d_out;
    float* part = (float*)d_ws;            // CRF_B/BT = 1024 floats

    const int nblocks = CRF_B / BT;        // 1024
    crf_fused_kernel<<<nblocks, 256, 0, stream>>>(X, Y, W, T, part);
    crf_reduce_kernel<<<1, 256, 0, stream>>>(part, out, nblocks);
}
